// SingleForgetGateTreeGRU_45664092291533
// MI455X (gfx1250) — matrix operand order repaired and checked by reading, not compiled, not run on hardware
//
#include <hip/hip_runtime.h>
#include <hip/hip_bf16.h>

// ---------------------------------------------------------------------------
// TreeGRU (4-ary, depth 9, H=512) on gfx1250, bf16 WMMA with f32 accumulate.
// Fast path: bf16 shadow state + pre-converted bf16 weights; 128x128 block
// tile, 8 waves each computing a 64x32 fragment block (8 WMMA / K-step),
// ping-pong LDS double-buffering with a single barrier per K-step.
// ---------------------------------------------------------------------------

typedef __attribute__((ext_vector_type(16))) __bf16 bf16x16;
typedef __attribute__((ext_vector_type(8)))  float  f32x8;

#define HDIM   512
#define NNODES 87381      // (4^9 - 1) / 3
#define BK     32
#define LDT    20         // LDS tile row stride in dwords (16 data + 4 pad)
// fast-path tile
#define BMF    128
#define BNF    128
// fallback tile
#define BM     64
#define BN     64

union Frag { bf16x16 v; uint4 q[2]; };

__device__ __forceinline__ unsigned int pack_bf2(float a, float b) {
  union { __bf16 h[2]; unsigned int u; } x;
  x.h[0] = (__bf16)a; x.h[1] = (__bf16)b;
  return x.u;
}
__device__ __forceinline__ unsigned short bf1(float a) {
  union { __bf16 h; unsigned short u; } x;
  x.h = (__bf16)a; return x.u;
}
__device__ __forceinline__ float bf2f(unsigned short u) {
  union { unsigned int u; float f; } x;
  x.u = ((unsigned int)u) << 16; return x.f;
}
__device__ __forceinline__ float sigmoidf_(float v) {
  return 1.0f / (1.0f + __expf(-v));
}

// ---------------- one-time fp32 -> bf16 conversion (8 elems / thread) ------
__global__ void cvt_bf16(const float* __restrict__ src,
                         unsigned short* __restrict__ dst, long long n8) {
  long long i = (long long)blockIdx.x * blockDim.x + threadIdx.x;
  if (i >= n8) return;
  const float4* s = (const float4*)(src + i * 8);
  float4 a = s[0], b = s[1];
  uint4 o;
  o.x = pack_bf2(a.x, a.y); o.y = pack_bf2(a.z, a.w);
  o.z = pack_bf2(b.x, b.y); o.w = pack_bf2(b.z, b.w);
  ((uint4*)dst)[i] = o;
}

// ---------------- rh = tile(r,4) (*) h_cat, bf16 out (8 elems / thread) ----
__global__ void mk_rh(const unsigned short* __restrict__ hc,
                      const float* __restrict__ r,
                      unsigned short* __restrict__ rh, long long n8) {
  long long i = (long long)blockIdx.x * blockDim.x + threadIdx.x;
  if (i >= n8) return;
  long long e   = i * 8;              // element index within [m * 2048)
  long long row = e >> 11;
  int       k   = (int)(e & 2047);
  union { uint4 q; unsigned short s[8]; } in, out;
  in.q = ((const uint4*)hc)[i];
  const float4* rp = (const float4*)(r + row * HDIM + (k & (HDIM - 1)));
  float4 r0 = rp[0], r1 = rp[1];
  out.s[0] = bf1(bf2f(in.s[0]) * r0.x);
  out.s[1] = bf1(bf2f(in.s[1]) * r0.y);
  out.s[2] = bf1(bf2f(in.s[2]) * r0.z);
  out.s[3] = bf1(bf2f(in.s[3]) * r0.w);
  out.s[4] = bf1(bf2f(in.s[4]) * r1.x);
  out.s[5] = bf1(bf2f(in.s[5]) * r1.y);
  out.s[6] = bf1(bf2f(in.s[6]) * r1.z);
  out.s[7] = bf1(bf2f(in.s[7]) * r1.w);
  ((uint4*)rh)[i] = out.q;
}

// ---------------- fast-path GEMM: A and W already bf16 ---------------------
// 256 threads, 128x128 tile, wave grid 2(M)x4(N), 64x32 frag block per wave.
// MODE 0: h = tanh(A @ W^T + b); also writes h_bf
// MODE 1: zr = A @ W^T + b; sigmoid -> z / r workspace
// MODE 2: cand = tanh(A @ W^T + b); h[s+row] = childsum*z + (1-z)*cand (+h_bf)
template<int MODE>
__global__ __launch_bounds__(256)
void gemm_bf(const unsigned short* __restrict__ A, int lda,
             const unsigned short* __restrict__ W, int K,
             const float* __restrict__ bias, int Mtot,
             float* __restrict__ zbuf, float* __restrict__ rbuf,
             float* __restrict__ h, unsigned short* __restrict__ hbf,
             long long s)
{
  __shared__ unsigned int As[2][BMF * LDT];
  __shared__ unsigned int Bs[2][BNF * LDT];

  const int t  = threadIdx.x;
  const int wv = t >> 5, ln = t & 31, lr = ln & 15, lg = ln >> 4;
  const int wm = wv & 1;        // M half: 64 rows
  const int wn = wv >> 1;       // N quarter: 32 cols
  const int mB = blockIdx.x * BMF, nB = blockIdx.y * BNF;
  const int rowS = t >> 1, hs = t & 1;

  f32x8 acc[4][2] = {};

  int grA = mB + rowS;
  if (MODE == 0 && grA >= Mtot) grA = Mtot - 1;   // clamp M tail (x bounds)
  const unsigned short* aRow = A + (size_t)grA * lda + hs * 16;
  const unsigned short* bRow = W + (size_t)(nB + rowS) * K + hs * 16;
  uint4* stA[2] = { (uint4*)&As[0][rowS * LDT + hs * 8],
                    (uint4*)&As[1][rowS * LDT + hs * 8] };
  uint4* stB[2] = { (uint4*)&Bs[0][rowS * LDT + hs * 8],
                    (uint4*)&Bs[1][rowS * LDT + hs * 8] };

  auto compute = [&](int b) {
    Frag bf0, bf1;
    // B (32x16): lanes 0-15 K={0..7,16..23}, lanes 16-31 K={8..15,24..31}
    const uint4* bb0 = (const uint4*)&Bs[b][(wn * 32      + lr) * LDT];
    const uint4* bb1 = (const uint4*)&Bs[b][(wn * 32 + 16 + lr) * LDT];
    bf0.q[0] = bb0[lg]; bf0.q[1] = bb0[lg + 2];
    bf1.q[0] = bb1[lg]; bf1.q[1] = bb1[lg + 2];
#pragma unroll
    for (int tm = 0; tm < 4; ++tm) {
      // A (16x32): lanes 0-15 K={0..7,16..23}, lanes 16-31 K={8..15,24..31}
      Frag af;
      const unsigned int* ar = &As[b][(wm * 64 + tm * 16 + lr) * LDT];
      af.q[0] = *(const uint4*)&ar[lg ? 4 : 0];
      af.q[1] = *(const uint4*)&ar[lg ? 12 : 8];
      acc[tm][0] = __builtin_amdgcn_wmma_f32_16x16x32_bf16(
          false, af.v, false, bf0.v, (short)0, acc[tm][0], false, false);
      acc[tm][1] = __builtin_amdgcn_wmma_f32_16x16x32_bf16(
          false, af.v, false, bf1.v, (short)0, acc[tm][1], false, false);
    }
  };

  // prologue: stage K-tile 0 into buffer 0
  {
    const uint4* ap = (const uint4*)aRow;
    const uint4* bp = (const uint4*)bRow;
    uint4 qa0 = ap[0], qa1 = ap[1], qb0 = bp[0], qb1 = bp[1];
    stA[0][0] = qa0; stA[0][1] = qa1;
    stB[0][0] = qb0; stB[0][1] = qb1;
  }
  __syncthreads();

  int buf = 0;
  for (int k0 = BK; k0 < K; k0 += BK) {
    // fetch next K-tile (latency overlapped with WMMA below)
    const uint4* ap = (const uint4*)(aRow + k0);
    const uint4* bp = (const uint4*)(bRow + k0);
    uint4 qa0 = ap[0], qa1 = ap[1], qb0 = bp[0], qb1 = bp[1];

    compute(buf);

    const int nb = buf ^ 1;
    stA[nb][0] = qa0; stA[nb][1] = qa1;
    stB[nb][0] = qb0; stB[nb][1] = qb1;
    __syncthreads();
    buf = nb;
  }
  compute(buf);

  // ---------------- fused epilogues ----------------------------------------
#pragma unroll
  for (int tn = 0; tn < 2; ++tn) {
    const int   col = nB + wn * 32 + tn * 16 + lr;
    const float bc  = bias[col];
#pragma unroll
    for (int tm = 0; tm < 4; ++tm) {
#pragma unroll
      for (int v = 0; v < 8; ++v) {
        const int row = mB + wm * 64 + tm * 16 + lg * 8 + v;
        if (row >= Mtot) continue;
        const float val = acc[tm][tn][v] + bc;
        if (MODE == 0) {
          const float o = tanhf(val);
          h  [(size_t)row * HDIM + col] = o;
          hbf[(size_t)row * HDIM + col] = bf1(o);
        } else if (MODE == 1) {
          const float sg = sigmoidf_(val);
          if (col < HDIM) zbuf[(size_t)row * HDIM + col] = sg;
          else            rbuf[(size_t)row * HDIM + (col - HDIM)] = sg;
        } else {
          const float  cand = tanhf(val);
          const size_t node = (size_t)s + (size_t)row;
          const float* ch   = h + (node * 4 + 1) * HDIM + col;
          const float  hsum = ch[0] + ch[HDIM] + ch[2 * HDIM] + ch[3 * HDIM];
          const float  z    = zbuf[(size_t)row * HDIM + col];
          const float  hn   = hsum * z + (1.0f - z) * cand;
          h  [node * HDIM + col] = hn;
          hbf[node * HDIM + col] = bf1(hn);
        }
      }
    }
  }
}

// ---------------- fallback GEMM (fp32 inputs, convert in loop) -------------
template<int MODE>
__global__ __launch_bounds__(128)
void gemm_f32(const float* __restrict__ A, int lda,
              const float* __restrict__ W, int K,
              const float* __restrict__ bias, int Mtot,
              float* __restrict__ zbuf, float* __restrict__ rbuf,
              float* __restrict__ h, long long s)
{
  __shared__ unsigned int As[BM * LDT];
  __shared__ unsigned int Bs[BN * LDT];
  const int t = threadIdx.x, wv = t >> 5, ln = t & 31, lr = ln & 15, lg = ln >> 4;
  const int mB = blockIdx.x * BM, nB = blockIdx.y * BN, rowS = t >> 1, hs = t & 1;
  f32x8 acc[4] = {};
  for (int k0 = 0; k0 < K; k0 += BK) {
    {
      int gr = mB + rowS;
      if (MODE == 0 && gr >= Mtot) gr = Mtot - 1;
      const float* ap = A + (size_t)gr * lda + (k0 + hs * 16);
      float4 f0 = ((const float4*)ap)[0], f1 = ((const float4*)ap)[1];
      float4 f2 = ((const float4*)ap)[2], f3 = ((const float4*)ap)[3];
      if (MODE == 2) {
        const float* rp = rbuf + (size_t)gr * HDIM + ((k0 + hs * 16) & (HDIM - 1));
        float4 r0 = ((const float4*)rp)[0], r1 = ((const float4*)rp)[1];
        float4 r2 = ((const float4*)rp)[2], r3 = ((const float4*)rp)[3];
        f0.x *= r0.x; f0.y *= r0.y; f0.z *= r0.z; f0.w *= r0.w;
        f1.x *= r1.x; f1.y *= r1.y; f1.z *= r1.z; f1.w *= r1.w;
        f2.x *= r2.x; f2.y *= r2.y; f2.z *= r2.z; f2.w *= r2.w;
        f3.x *= r3.x; f3.y *= r3.y; f3.z *= r3.z; f3.w *= r3.w;
      }
      unsigned int* d = &As[rowS * LDT + hs * 8];
      d[0] = pack_bf2(f0.x, f0.y); d[1] = pack_bf2(f0.z, f0.w);
      d[2] = pack_bf2(f1.x, f1.y); d[3] = pack_bf2(f1.z, f1.w);
      d[4] = pack_bf2(f2.x, f2.y); d[5] = pack_bf2(f2.z, f2.w);
      d[6] = pack_bf2(f3.x, f3.y); d[7] = pack_bf2(f3.z, f3.w);
    }
    {
      const float* bp = W + (size_t)(nB + rowS) * K + (k0 + hs * 16);
      float4 f0 = ((const float4*)bp)[0], f1 = ((const float4*)bp)[1];
      float4 f2 = ((const float4*)bp)[2], f3 = ((const float4*)bp)[3];
      unsigned int* d = &Bs[rowS * LDT + hs * 8];
      d[0] = pack_bf2(f0.x, f0.y); d[1] = pack_bf2(f0.z, f0.w);
      d[2] = pack_bf2(f1.x, f1.y); d[3] = pack_bf2(f1.z, f1.w);
      d[4] = pack_bf2(f2.x, f2.y); d[5] = pack_bf2(f2.z, f2.w);
      d[6] = pack_bf2(f3.x, f3.y); d[7] = pack_bf2(f3.z, f3.w);
    }
    __syncthreads();
    Frag bfr;
    {
      const uint4* bb = (const uint4*)&Bs[(wv * 16 + lr) * LDT];
      bfr.q[0] = bb[lg];
      bfr.q[1] = bb[lg + 2];
    }
#pragma unroll
    for (int tm = 0; tm < 4; ++tm) {
      Frag afr;
      const unsigned int* ar = &As[(tm * 16 + lr) * LDT];
      afr.q[0] = *(const uint4*)&ar[lg ? 4 : 0];
      afr.q[1] = *(const uint4*)&ar[lg ? 12 : 8];
      acc[tm] = __builtin_amdgcn_wmma_f32_16x16x32_bf16(
          false, afr.v, false, bfr.v, (short)0, acc[tm], false, false);
    }
    __syncthreads();
  }
  const int col = nB + wv * 16 + lr;
  const float bc = bias[col];
#pragma unroll
  for (int tm = 0; tm < 4; ++tm) {
#pragma unroll
    for (int v = 0; v < 8; ++v) {
      const int row = mB + tm * 16 + lg * 8 + v;
      if (row >= Mtot) continue;
      const float val = acc[tm][v] + bc;
      if (MODE == 0) {
        h[(size_t)row * HDIM + col] = tanhf(val);
      } else if (MODE == 1) {
        const float sg = sigmoidf_(val);
        if (col < HDIM) zbuf[(size_t)row * HDIM + col] = sg;
        else            rbuf[(size_t)row * HDIM + (col - HDIM)] = sg;
      } else {
        const float  cand = tanhf(val);
        const size_t node = (size_t)s + (size_t)row;
        const float* ch   = h + (node * 4 + 1) * HDIM + col;
        const float  hsum = ch[0] + ch[HDIM] + ch[2 * HDIM] + ch[3 * HDIM];
        const float  z    = zbuf[(size_t)row * HDIM + col];
        h[node * HDIM + col] = hsum * z + (1.0f - z) * cand;
      }
    }
  }
}

extern "C" void kernel_launch(void* const* d_in, const int* in_sizes, int n_in,
                              void* d_out, int out_size, void* d_ws, size_t ws_size,
                              hipStream_t stream)
{
  const float* x     = (const float*)d_in[0];
  const float* Ww    = (const float*)d_in[1];
  const float* Wb    = (const float*)d_in[2];
  const float* Uzr_w = (const float*)d_in[3];
  const float* Uzr_b = (const float*)d_in[4];
  const float* Uh_w  = (const float*)d_in[5];
  const float* Uh_b  = (const float*)d_in[6];
  (void)in_sizes; (void)n_in; (void)out_size;

  float* h = (float*)d_out;                       // [87381, 512] fp32

  // ---- workspace layout (fast path) ----
  const size_t b_hbf = (size_t)NNODES * HDIM * 2;        // 89.5 MB
  const size_t b_z   = (size_t)16384 * HDIM * 4;         // 33.5 MB
  const size_t b_r   = b_z;                              // 33.5 MB
  const size_t b_rh  = (size_t)16384 * 4 * HDIM * 2;     // 67.1 MB
  const size_t b_ww  = (size_t)HDIM * HDIM * 2;
  const size_t b_uzr = (size_t)(2 * HDIM) * (4 * HDIM) * 2;
  const size_t b_uh  = (size_t)HDIM * (4 * HDIM) * 2;
  const size_t need  = b_hbf + b_z + b_r + b_rh + b_ww + b_uzr + b_uh;

  char* p = (char*)d_ws;
  unsigned short* hbf  = (unsigned short*)p;               p += b_hbf;
  float*          zbuf = (float*)p;                        p += b_z;
  float*          rbuf = (float*)p;                        p += b_r;
  unsigned short* rh   = (unsigned short*)p;               p += b_rh;
  unsigned short* wwb  = (unsigned short*)p;               p += b_ww;
  unsigned short* uzrb = (unsigned short*)p;               p += b_uzr;
  unsigned short* uhb  = (unsigned short*)p;
  // x_bf reuses the (temporally dead) z/r/rh region: 89.5 MB <= 134 MB
  unsigned short* xbf  = (unsigned short*)zbuf;

  if (ws_size >= need) {
    dim3 blk(256);
    // ---- one-time conversions ----
    {
      long long n8;
      n8 = (long long)NNODES * HDIM / 8;
      cvt_bf16<<<dim3((unsigned)((n8 + 255) / 256)), dim3(256), 0, stream>>>(x, xbf, n8);
      n8 = (long long)HDIM * HDIM / 8;
      cvt_bf16<<<dim3((unsigned)((n8 + 255) / 256)), dim3(256), 0, stream>>>(Ww, wwb, n8);
      n8 = (long long)(2 * HDIM) * (4 * HDIM) / 8;
      cvt_bf16<<<dim3((unsigned)((n8 + 255) / 256)), dim3(256), 0, stream>>>(Uzr_w, uzrb, n8);
      n8 = (long long)HDIM * (4 * HDIM) / 8;
      cvt_bf16<<<dim3((unsigned)((n8 + 255) / 256)), dim3(256), 0, stream>>>(Uh_w, uhb, n8);
    }
    // ---- leaf transform ----
    gemm_bf<0><<<dim3((NNODES + BMF - 1) / BMF, HDIM / BNF), blk, 0, stream>>>(
        xbf, HDIM, wwb, HDIM, Wb, NNODES, nullptr, nullptr, h, hbf, 0);
    // ---- internal levels, deepest first ----
    for (int l = 7; l >= 0; --l) {
      const long long m  = 1LL << (2 * l);
      const long long sI = ((1LL << (2 * l)) - 1) / 3;
      const unsigned short* hcb = hbf + (size_t)(4 * sI + 1) * HDIM;
      const unsigned gm = (unsigned)((m + BMF - 1) / BMF);

      gemm_bf<1><<<dim3(gm, (2 * HDIM) / BNF), blk, 0, stream>>>(
          hcb, 4 * HDIM, uzrb, 4 * HDIM, Uzr_b, (int)m, zbuf, rbuf, h, hbf, sI);

      const long long n8 = m * (4 * HDIM) / 8;
      mk_rh<<<dim3((unsigned)((n8 + 255) / 256)), dim3(256), 0, stream>>>(hcb, rbuf, rh, n8);

      gemm_bf<2><<<dim3(gm, HDIM / BNF), blk, 0, stream>>>(
          rh, 4 * HDIM, uhb, 4 * HDIM, Uh_b, (int)m, zbuf, rbuf, h, hbf, sI);
    }
  } else {
    // ---- fallback: convert-in-loop (needs only z + r = 67 MB) ----
    dim3 blk(128);
    float* zb = (float*)d_ws;
    float* rb = zb + (size_t)16384 * HDIM;
    gemm_f32<0><<<dim3((NNODES + BM - 1) / BM, HDIM / BN), blk, 0, stream>>>(
        x, HDIM, Ww, HDIM, Wb, NNODES, nullptr, nullptr, h, 0);
    for (int l = 7; l >= 0; --l) {
      const long long m  = 1LL << (2 * l);
      const long long sI = ((1LL << (2 * l)) - 1) / 3;
      const float* Hc = h + (size_t)(4 * sI + 1) * HDIM;
      const unsigned gm = (unsigned)((m + BM - 1) / BM);
      gemm_f32<1><<<dim3(gm, (2 * HDIM) / BN), blk, 0, stream>>>(
          Hc, 4 * HDIM, Uzr_w, 4 * HDIM, Uzr_b, (int)m, zb, rb, h, sI);
      gemm_f32<2><<<dim3(gm, HDIM / BN), blk, 0, stream>>>(
          Hc, 4 * HDIM, Uh_w, 4 * HDIM, Uh_b, (int)m, zb, rb, h, sI);
    }
  }
}
